// TP_lstm_19018115186832
// MI455X (gfx1250) — compile-verified, matrix operand order checked by reading, not run on hardware
//
#include <hip/hip_runtime.h>
#include <hip/hip_bf16.h>

// ---------------------------------------------------------------------------
// Persistent bf16-WMMA LSTM for MI455X (gfx1250, wave32).
//   grid = 32 blocks, each owns 16 batch rows for all 1024 time steps.
//   Weights live in VGPRs as WMMA B-fragments (loop-invariant).
//   LDS (< 64KB) used for h staging + fp32 gate exchange between waves.
// ---------------------------------------------------------------------------

typedef __attribute__((ext_vector_type(16))) __bf16 v16bf;
typedef __attribute__((ext_vector_type(8)))  float  v8f;

#define S_    512
#define T_    1024
#define I_    3
#define H_    128
#define G4_   512     // 4*H
#define SPIN_ 256     // 2*H
#define OH_   64
#define OUTD_ 3
#define TP_   4
#define ROWS_ 16      // batch rows per block
#define BLOCKT 256    // 8 wave32

union FragBF { v16bf v; unsigned u[8]; };
union FragF  { v8f   v; float    f[8]; };

__device__ __forceinline__ unsigned pack_bf2(float lo, float hi) {
    __bf16 a = (__bf16)lo, b = (__bf16)hi;
    unsigned short ua = __builtin_bit_cast(unsigned short, a);
    unsigned short ub = __builtin_bit_cast(unsigned short, b);
    return (unsigned)ua | ((unsigned)ub << 16);
}

// K offset inside a 32-wide k-tile for vgpr v / lane-half (ISA 16-bit layout)
__device__ __forceinline__ int kofs(int v, int half) {
    return 2*(v & 3) + 8*half + 16*(v >> 2);
}

// B tile (32x16): B[k][n] = W[n][k]; W is row-major f32 with row stride ldk.
__device__ __forceinline__ FragBF load_b_tile(const float* __restrict__ W,
                                              int ldk, int n0, int k0, int lane) {
    FragBF b;
    int half = lane >> 4;
    int n = n0 + (lane & 15);
    #pragma unroll
    for (int v = 0; v < 8; ++v) {
        int k = k0 + kofs(v, half);
        b.u[v] = pack_bf2(W[n*ldk + k], W[n*ldk + k + 1]);
    }
    return b;
}

// A fragment (16x32) from LDS bf16 row-major [m][ld] (k pairs are contiguous).
__device__ __forceinline__ FragBF load_a_frag(const __bf16* sA, int ld, int k0, int lane) {
    FragBF a;
    int half = lane >> 4;
    int m = lane & 15;
    const unsigned* p = (const unsigned*)sA;
    #pragma unroll
    for (int v = 0; v < 8; ++v) {
        int k = k0 + kofs(v, half);          // even -> 4B aligned
        a.u[v] = p[(m*ld + k) >> 1];
    }
    return a;
}

// Scatter 16x16 f32 accumulator (C/D layout) into LDS row-major [m][ld].
__device__ __forceinline__ void store_acc(float* sC, int ld, int n0, int lane,
                                          const FragF& acc) {
    int half = lane >> 4;
    int n = n0 + (lane & 15);
    #pragma unroll
    for (int v = 0; v < 8; ++v) {
        int m = v + 8*half;
        sC[m*ld + n] = acc.f[v];
    }
}

__device__ __forceinline__ float sigf(float x) { return 1.0f / (1.0f + __expf(-x)); }

__global__ __launch_bounds__(BLOCKT)
void lstm_persistent_kernel(const float* __restrict__ input,
                            const float* __restrict__ lane_in,
                            const float* __restrict__ W_ih, const float* __restrict__ b_ih,
                            const float* __restrict__ W_hh, const float* __restrict__ b_hh,
                            const float* __restrict__ W_sp, const float* __restrict__ b_sp,
                            const float* __restrict__ lg_w1, const float* __restrict__ lg_b1,
                            const float* __restrict__ lg_w2, const float* __restrict__ lg_b2,
                            const float* __restrict__ out_w1, const float* __restrict__ out_b1,
                            const float* __restrict__ out_w2, const float* __restrict__ out_b2,
                            float* __restrict__ out)
{
    __shared__ __bf16 sH[ROWS_*H_];        //  4 KB  current h (A source, bf16)
    __shared__ __bf16 sGin[ROWS_*SPIN_];   //  8 KB  [h_sp | h_sm] (A source)
    __shared__ float  sG1[ROWS_*G4_];      // 32 KB  gate pre-activations (i,f,g,o)
    __shared__ float  sG2[ROWS_*H_];       //  8 KB  spatial-gate pre-activations
    __shared__ float  sO1[ROWS_*OH_];      //  4 KB  output-head hidden
    __shared__ float  sBih[G4_];           //  2 KB  b_ih + b_hh
    __shared__ __bf16 sWih[G4_*I_];        //  3 KB  W_ih (tiny K=3 handled scalar)
    __shared__ float  sBsp[H_];
    __shared__ float  sOb1[OH_];
    __shared__ float  sOw2[OUTD_*OH_];
    __shared__ float  sOb2[OUTD_];
    __shared__ float  sLc;
    // total ~62.6 KB

    const int tid  = threadIdx.x;
    const int w    = tid >> 5;
    const int lane = tid & 31;
    const int rowBase = blockIdx.x * ROWS_;

    // ----- small LDS tables ------------------------------------------------
    for (int n = tid; n < G4_;       n += BLOCKT) sBih[n] = b_ih[n] + b_hh[n];
    for (int e = tid; e < G4_*I_;    e += BLOCKT) sWih[e] = (__bf16)W_ih[e];
    for (int e = tid; e < H_;        e += BLOCKT) sBsp[e] = b_sp[e];
    for (int e = tid; e < OH_;       e += BLOCKT) sOb1[e] = out_b1[e];
    for (int e = tid; e < OUTD_*OH_; e += BLOCKT) sOw2[e] = out_w2[e];
    if (tid < OUTD_) sOb2[tid] = out_b2[tid];
    for (int e = tid; e < ROWS_*H_;    e += BLOCKT) sH[e]   = (__bf16)0.0f;
    for (int e = tid; e < ROWS_*SPIN_; e += BLOCKT) sGin[e] = (__bf16)0.0f;
    if (tid == 0) {  // lane-gate scalar MLP (LG=16), identical for every block
        float l = lane_in[0];
        float acc = lg_b2[0];
        #pragma unroll
        for (int k = 0; k < 16; ++k) {
            float tv = l * lg_w1[k] + lg_b1[k];
            acc += lg_w2[k] * (tv > 0.f ? tv : 0.f);
        }
        sLc = acc;
    }

    // ----- loop-invariant weight B-fragments, pinned in VGPRs --------------
    FragBF Bhh[16];                       // gates: 4 n-tiles x 4 k-steps / wave
    #pragma unroll
    for (int a = 0; a < 4; ++a)
        #pragma unroll
        for (int kt = 0; kt < 4; ++kt)
            Bhh[a*4 + kt] = load_b_tile(W_hh, H_, (4*w + a)*16, kt*32, lane);

    FragBF Bsp[8];                        // spatial: 1 n-tile x 8 k-steps / wave
    #pragma unroll
    for (int kt = 0; kt < 8; ++kt)
        Bsp[kt] = load_b_tile(W_sp, SPIN_, w*16, kt*32, lane);

    FragBF Bo1[4] = {};                   // out head: waves 0..3 only
    if (w < 4) {
        #pragma unroll
        for (int kt = 0; kt < 4; ++kt)
            Bo1[kt] = load_b_tile(out_w1, H_, w*16, kt*32, lane);
    }

    // ----- per-thread persistent cell state (c), fixed (m,j) ownership -----
    float c[8];
    #pragma unroll
    for (int q = 0; q < 8; ++q) c[q] = 0.f;
    const int m_e  = tid >> 4;            // elementwise row 0..15
    const int jb_e = (tid & 15) * 8;      // elementwise col base
    const int srow = rowBase + m_e;
    const bool lastRow  = (srow == S_ - 1);
    const bool firstRow = (srow == 0);
    const float* xrow = input + (size_t)srow * T_ * I_;

    __syncthreads();

    for (int t = 0; t < T_; ++t) {
        // ---- GEMM 1: gates = h @ W_hh^T  (M16 N512 K128) -------------------
        {
            FragBF A[4];
            #pragma unroll
            for (int kt = 0; kt < 4; ++kt) A[kt] = load_a_frag(sH, H_, kt*32, lane);
            #pragma unroll
            for (int a = 0; a < 4; ++a) {
                FragF acc;
                #pragma unroll
                for (int q = 0; q < 8; ++q) acc.f[q] = 0.f;
                #pragma unroll
                for (int kt = 0; kt < 4; ++kt)
                    acc.v = __builtin_amdgcn_wmma_f32_16x16x32_bf16(
                        false, A[kt].v, false, Bhh[a*4 + kt].v,
                        (short)0, acc.v, false, false);
                store_acc(sG1, G4_, (4*w + a)*16, lane, acc);
            }
        }
        // ---- GEMM 2: spatial = [h_sp|h_sm] @ W_sp^T  (M16 N128 K256) -------
        {
            FragF acc;
            #pragma unroll
            for (int q = 0; q < 8; ++q) acc.f[q] = 0.f;
            #pragma unroll
            for (int kt = 0; kt < 8; ++kt) {
                FragBF A = load_a_frag(sGin, SPIN_, kt*32, lane);
                acc.v = __builtin_amdgcn_wmma_f32_16x16x32_bf16(
                    false, A.v, false, Bsp[kt].v, (short)0, acc.v, false, false);
            }
            store_acc(sG2, H_, w*16, lane, acc);
        }
        __syncthreads();

        // ---- elementwise LSTM cell update (fp32) ---------------------------
        {
            const float lc = sLc;
            const float* xp = xrow + (size_t)t * I_;
            float x0 = xp[0]*lc, x1 = xp[1]*lc, x2 = xp[2]*lc;
            if (t + 8 < T_) __builtin_prefetch(xrow + (size_t)(t + 8)*I_, 0, 1);
            #pragma unroll
            for (int q = 0; q < 8; ++q) {
                int j = jb_e + q;
                float xi = x0*(float)sWih[(j      )*3+0] + x1*(float)sWih[(j      )*3+1] + x2*(float)sWih[(j      )*3+2];
                float xf = x0*(float)sWih[(128+j  )*3+0] + x1*(float)sWih[(128+j  )*3+1] + x2*(float)sWih[(128+j  )*3+2];
                float xg = x0*(float)sWih[(256+j  )*3+0] + x1*(float)sWih[(256+j  )*3+1] + x2*(float)sWih[(256+j  )*3+2];
                float xo = x0*(float)sWih[(384+j  )*3+0] + x1*(float)sWih[(384+j  )*3+1] + x2*(float)sWih[(384+j  )*3+2];
                float iv = sG1[m_e*G4_ +        j] + sBih[      j] + xi;
                float fv = sG1[m_e*G4_ + 128 +  j] + sBih[128 + j] + xf;
                float gv = sG1[m_e*G4_ + 256 +  j] + sBih[256 + j] + xg;
                float ov = sG1[m_e*G4_ + 384 +  j] + sBih[384 + j] + xo;
                float sp = sigf(sG2[m_e*H_ + j] + sBsp[j]);
                float cn = sigf(fv)*c[q] + sigf(iv)*tanhf(gv);
                float hn = sigf(ov)*tanhf(cn)*sp;
                c[q] = cn;
                sH[m_e*H_ + j]              = (__bf16)hn;
                sGin[m_e*SPIN_ + j]         = (__bf16)(lastRow  ? 0.f : hn);
                sGin[m_e*SPIN_ + 128 + j]   = (__bf16)(firstRow ? 0.f : hn);
            }
        }
        __syncthreads();

        // ---- GEMM 3: o1 = h_new @ out_w1^T  (M16 N64 K128), waves 0..3 -----
        if (w < 4) {
            FragF acc;
            #pragma unroll
            for (int q = 0; q < 8; ++q) acc.f[q] = 0.f;
            #pragma unroll
            for (int kt = 0; kt < 4; ++kt) {
                FragBF A = load_a_frag(sH, H_, kt*32, lane);
                acc.v = __builtin_amdgcn_wmma_f32_16x16x32_bf16(
                    false, A.v, false, Bo1[kt].v, (short)0, acc.v, false, false);
            }
            store_acc(sO1, OH_, w*16, lane, acc);
        }
        __syncthreads();

        // ---- tiny output head: y = relu(o1)@out_w2^T + b -------------------
        if (t >= TP_ && tid < ROWS_*OUTD_) {
            int m = tid / OUTD_, o = tid % OUTD_;
            float acc = sOb2[o];
            #pragma unroll 8
            for (int j = 0; j < OH_; ++j) {
                float r = sO1[m*OH_ + j] + sOb1[j];
                acc += (r > 0.f ? r : 0.f) * sOw2[o*OH_ + j];
            }
            out[((size_t)(rowBase + m) * (T_ - TP_) + (t - TP_)) * OUTD_ + o] = acc;
        }
        __syncthreads();
    }
}

extern "C" void kernel_launch(void* const* d_in, const int* in_sizes, int n_in,
                              void* d_out, int out_size, void* d_ws, size_t ws_size,
                              hipStream_t stream) {
    (void)in_sizes; (void)n_in; (void)out_size; (void)d_ws; (void)ws_size;
    const float* input  = (const float*)d_in[0];
    const float* lane   = (const float*)d_in[1];
    const float* W_ih   = (const float*)d_in[2];
    const float* b_ih   = (const float*)d_in[3];
    const float* W_hh   = (const float*)d_in[4];
    const float* b_hh   = (const float*)d_in[5];
    const float* W_sp   = (const float*)d_in[6];
    const float* b_sp   = (const float*)d_in[7];
    const float* lg_w1  = (const float*)d_in[8];
    const float* lg_b1  = (const float*)d_in[9];
    const float* lg_w2  = (const float*)d_in[10];
    const float* lg_b2  = (const float*)d_in[11];
    const float* out_w1 = (const float*)d_in[12];
    const float* out_b1 = (const float*)d_in[13];
    const float* out_w2 = (const float*)d_in[14];
    const float* out_b2 = (const float*)d_in[15];
    float* out = (float*)d_out;

    dim3 grid(S_ / ROWS_);   // 32 persistent blocks
    dim3 block(BLOCKT);      // 8 wave32
    lstm_persistent_kernel<<<grid, block, 0, stream>>>(
        input, lane, W_ih, b_ih, W_hh, b_hh, W_sp, b_sp,
        lg_w1, lg_b1, lg_w2, lg_b2, out_w1, out_b1, out_w2, out_b2, out);
}
